// PointMAE_32976758898982
// MI455X (gfx1250) — matrix operand order repaired and checked by reading, not compiled, not run on hardware
//
#include <hip/hip_runtime.h>
#include <math.h>

// ---------------- problem constants (from reference) ----------------
#define BB     64
#define NN     4096
#define CIN    6
#define NPATCH 128
#define PPTS   32
#define DD     384
#define HH     6
#define DHEAD  64
#define DFF    1536
#define NMASK  96
#define NVIS   32
#define ENC_L  12
#define DEC_L  4

typedef __attribute__((ext_vector_type(16))) _Float16 v16h;
typedef __attribute__((ext_vector_type(4)))  _Float16 v4h;
typedef __attribute__((ext_vector_type(8)))  float    v8f;

// =====================================================================
// Tiled WMMA TN-GEMM:  C[M,N] = act( A[M,K] @ W[N,K]^T + bias + res )
// 64x64 output tile per 256-thread block (8 waves). A/W K-slabs staged
// through LDS as f16 (converted once, shared by all 8 waves: 4x less
// global traffic per operand). Staging is vectorized: float4 global
// loads (b128) -> packed f16 -> 8B LDS stores (b64). Each wave owns two
// 16x16 accumulators -> two back-to-back v_wmma per K step.
// Requires M%64==0, N%64==0, K%32==0, 16B-aligned A/W.
// =====================================================================
__global__ __launch_bounds__(256) void k_gemm_wmma_t(
    const float* __restrict__ A, const float* __restrict__ W,
    const float* __restrict__ bias, const float* __restrict__ res,
    float* __restrict__ Cm, int M, int N, int K, int act)
{
  __shared__ _Float16 lA[64 * 32];
  __shared__ _Float16 lW[64 * 32];
  const int tid  = threadIdx.x;
  const int wv   = tid >> 5;          // wave 0..7
  const int lane = tid & 31;
  const int l15  = lane & 15;
  const int hi   = lane >> 4;         // K-half carried by this lane
  const int ti0  = wv >> 2;           // M sub-tile 0..1 (second at +32)
  const int tj   = wv & 3;            // N sub-tile 0..3
  const int bm0  = blockIdx.y * 64;
  const int bn0  = blockIdx.x * 64;

  v8f acc0 = {}, acc1 = {};
  for (int k0 = 0; k0 < K; k0 += 32) {
    __syncthreads();
    // cooperative stage: 64 rows x 32 K of A and W, f32 -> f16,
    // two float4 chunks per thread per operand (512 chunks total).
#pragma unroll
    for (int u = 0; u < 2; ++u) {
      const int i  = tid + u * 256;        // chunk id 0..511
      const int r  = i >> 3;               // row 0..63
      const int c4 = (i & 7) << 2;         // K offset 0,4,..,28
      const float4 a4 = *(const float4*)&A[(size_t)(bm0 + r) * K + k0 + c4];
      const float4 w4 = *(const float4*)&W[(size_t)(bn0 + r) * K + k0 + c4];
      v4h ah, wh;
      ah[0] = (_Float16)a4.x; ah[1] = (_Float16)a4.y;
      ah[2] = (_Float16)a4.z; ah[3] = (_Float16)a4.w;
      wh[0] = (_Float16)w4.x; wh[1] = (_Float16)w4.y;
      wh[2] = (_Float16)w4.z; wh[3] = (_Float16)w4.w;
      *(v4h*)&lA[r * 32 + c4] = ah;
      *(v4h*)&lW[r * 32 + c4] = wh;
      if (k0 + 32 < K) {                   // stream next K-slab toward L2
        __builtin_prefetch((const void*)&A[(size_t)(bm0 + r) * K + k0 + 32 + c4], 0, 1);
        __builtin_prefetch((const void*)&W[(size_t)(bn0 + r) * K + k0 + 32 + c4], 0, 1);
      }
    }
    __syncthreads();
    // B fragment for this wave's N sub-tile (ISA 7.12.2 16-bit layouts)
    v16h bv;
    const int wrow = tj * 16 + l15;
#pragma unroll
    for (int j = 0; j < 16; ++j) bv[j] = lW[wrow * 32 + hi * 8 + (j < 8 ? j : j + 8)];
    // A fragments for the two M sub-tiles
    v16h a0, a1;
    const int ar0 = ti0 * 16 + l15;
    const int ar1 = ar0 + 32;
#pragma unroll
    for (int j = 0; j < 8; ++j) {
      a0[j]     = lA[ar0 * 32 + hi * 8 + j];
      a0[8 + j] = lA[ar0 * 32 + 16 + hi * 8 + j];
      a1[j]     = lA[ar1 * 32 + hi * 8 + j];
      a1[8 + j] = lA[ar1 * 32 + 16 + hi * 8 + j];
    }
    acc0 = __builtin_amdgcn_wmma_f32_16x16x32_f16(false, a0, false, bv,
                                                  (short)0, acc0, false, false);
    acc1 = __builtin_amdgcn_wmma_f32_16x16x32_f16(false, a1, false, bv,
                                                  (short)0, acc1, false, false);
  }
  // epilogue (D layout: VGPR i -> M = hi*8 + i, N = lane&15 within tile)
  const int   cn   = bn0 + tj * 16 + l15;
  const float bval = bias ? bias[cn] : 0.f;
#pragma unroll
  for (int i = 0; i < 8; ++i) {
    int rm = bm0 + ti0 * 16 + hi * 8 + i;
    float v = acc0[i] + bval;
    if (res) v += res[(size_t)rm * N + cn];
    if (act) v = fmaxf(v, 0.f);
    Cm[(size_t)rm * N + cn] = v;
    rm += 32;
    v = acc1[i] + bval;
    if (res) v += res[(size_t)rm * N + cn];
    if (act) v = fmaxf(v, 0.f);
    Cm[(size_t)rm * N + cn] = v;
  }
}

// =====================================================================
// Fallback WMMA GEMM (one wave per 16x16 tile) for N not multiple of 64
// (only the 96-wide head projection uses this).
// =====================================================================
__global__ __launch_bounds__(32) void k_gemm_wmma(
    const float* __restrict__ A, const float* __restrict__ W,
    const float* __restrict__ bias, const float* __restrict__ res,
    float* __restrict__ Cm, int M, int N, int K, int act)
{
  const int tn   = blockIdx.x;
  const int tm   = blockIdx.y;
  const int lane = threadIdx.x;
  const int l15  = lane & 15;
  const int hi   = lane >> 4;
  const float* Ar = A + (size_t)(tm * 16 + l15) * K;
  const float* Wr = W + (size_t)(tn * 16 + l15) * K + hi * 8;

  v8f acc = {};
  for (int k0 = 0; k0 < K; k0 += 32) {
    v16h av, bv;
    const float* ap = Ar + k0 + hi * 8;
#pragma unroll
    for (int j = 0; j < 8; ++j)  av[j]     = (_Float16)ap[j];
#pragma unroll
    for (int j = 0; j < 8; ++j)  av[8 + j] = (_Float16)ap[16 + j];
    const float* wp = Wr + k0;
#pragma unroll
    for (int j = 0; j < 16; ++j) bv[j]     = (_Float16)wp[j < 8 ? j : j + 8];
    if (k0 + 32 < K) {
      __builtin_prefetch((const void*)(ap + 32), 0, 1);
      __builtin_prefetch((const void*)(wp + 32), 0, 1);
    }
    acc = __builtin_amdgcn_wmma_f32_16x16x32_f16(false, av, false, bv,
                                                 (short)0, acc, false, false);
  }
  const int   cn    = tn * 16 + l15;
  const int   rbase = tm * 16 + hi * 8;
  const float bval  = bias ? bias[cn] : 0.f;
#pragma unroll
  for (int i = 0; i < 8; ++i) {
    const int rm = rbase + i;
    float v = acc[i] + bval;
    if (res) v += res[(size_t)rm * N + cn];
    if (act) v = fmaxf(v, 0.f);
    Cm[(size_t)rm * N + cn] = v;
  }
}

// =====================================================================
// Farthest point sampling: one block per batch, 128 dependent rounds.
// Tie-breaking = lowest global index (matches jnp.argmax).
// =====================================================================
__global__ __launch_bounds__(256) void k_fps(const float* __restrict__ x,
                                             int* __restrict__ fpsI,
                                             float* __restrict__ centers)
{
  const int b   = blockIdx.x;
  const int tid = threadIdx.x;                 // 256 threads, 16 pts each
  const float* xb = x + (size_t)b * NN * CIN;
  float dl[16];
#pragma unroll
  for (int k = 0; k < 16; ++k) dl[k] = 1e10f;

  __shared__ float cen[3];
  __shared__ float rv[256];
  __shared__ int   ri[256];
  __shared__ int   s_far;
  if (tid == 0) s_far = 0;
  __syncthreads();

  for (int it = 0; it < NPATCH; ++it) {
    const int far = s_far;
    if (tid == 0) {
      fpsI[b * NPATCH + it] = far;
      const float* pf = xb + (size_t)far * CIN;
      centers[(b * NPATCH + it) * 3 + 0] = pf[0];
      centers[(b * NPATCH + it) * 3 + 1] = pf[1];
      centers[(b * NPATCH + it) * 3 + 2] = pf[2];
      cen[0] = pf[0]; cen[1] = pf[1]; cen[2] = pf[2];
    }
    __syncthreads();
    const float c0 = cen[0], c1 = cen[1], c2 = cen[2];
    float bv = -1.f; int bi = 0;
#pragma unroll
    for (int k = 0; k < 16; ++k) {
      const int n = k * 256 + tid;
      const float dx = xb[(size_t)n * CIN + 0] - c0;
      const float dy = xb[(size_t)n * CIN + 1] - c1;
      const float dz = xb[(size_t)n * CIN + 2] - c2;
      const float d  = dx * dx + dy * dy + dz * dz;
      dl[k] = fminf(dl[k], d);
      if (dl[k] > bv) { bv = dl[k]; bi = n; }
    }
    rv[tid] = bv; ri[tid] = bi;
    __syncthreads();
    for (int s2 = 128; s2 > 0; s2 >>= 1) {
      if (tid < s2) {
        const float v2 = rv[tid + s2]; const int i2 = ri[tid + s2];
        if (v2 > rv[tid] || (v2 == rv[tid] && i2 < ri[tid])) { rv[tid] = v2; ri[tid] = i2; }
      }
      __syncthreads();
    }
    if (tid == 0) s_far = ri[0];
    __syncthreads();
  }
}

// =====================================================================
// kNN (ball-query via top-32 smallest sq-distance): block per (b,m).
// =====================================================================
__global__ __launch_bounds__(256) void k_knn(const float* __restrict__ x,
                                             const float* __restrict__ centers,
                                             int* __restrict__ knnI)
{
  const int bm  = blockIdx.x;
  const int b   = bm / NPATCH;
  const int tid = threadIdx.x;
  __shared__ float sd[NN];
  __shared__ float rv[256];
  __shared__ int   ri[256];
  const float* xb = x + (size_t)b * NN * CIN;
  const float c0 = centers[bm * 3 + 0];
  const float c1 = centers[bm * 3 + 1];
  const float c2 = centers[bm * 3 + 2];
#pragma unroll
  for (int k = 0; k < 16; ++k) {
    const int n = k * 256 + tid;
    const float dx = xb[(size_t)n * CIN + 0] - c0;
    const float dy = xb[(size_t)n * CIN + 1] - c1;
    const float dz = xb[(size_t)n * CIN + 2] - c2;
    sd[n] = dx * dx + dy * dy + dz * dz;
  }
  __syncthreads();
  for (int j = 0; j < PPTS; ++j) {
    float bv = 3.4e38f; int bi = 0;
#pragma unroll
    for (int k = 0; k < 16; ++k) {
      const int n = k * 256 + tid;
      const float v = sd[n];
      if (v < bv) { bv = v; bi = n; }
    }
    rv[tid] = bv; ri[tid] = bi;
    __syncthreads();
    for (int s2 = 128; s2 > 0; s2 >>= 1) {
      if (tid < s2) {
        const float v2 = rv[tid + s2]; const int i2 = ri[tid + s2];
        if (v2 < rv[tid] || (v2 == rv[tid] && i2 < ri[tid])) { rv[tid] = v2; ri[tid] = i2; }
      }
      __syncthreads();
    }
    if (tid == 0) { knnI[(size_t)bm * PPTS + j] = ri[0]; sd[ri[0]] = 3.4e38f; }
    __syncthreads();
  }
}

// ---------------- patch gather helper (into LDS) ----------------
__device__ inline void gather_feat(const float* __restrict__ x,
                                   const int* __restrict__ knnI,
                                   const float* __restrict__ centers,
                                   int bm, int b, int tid, float* feat)
{
  for (int i = tid; i < PPTS * CIN; i += 128) {
    const int p = i / CIN, c = i % CIN;
    const int n = knnI[(size_t)bm * PPTS + p];
    float v = x[((size_t)b * NN + n) * CIN + c];
    if (c < 3) v -= centers[bm * 3 + c];
    feat[i] = v;
  }
}

// conv1 raw stats (deterministic per-block partials)
__global__ __launch_bounds__(128) void k_pstats1(
    const float* __restrict__ x, const int* __restrict__ knnI,
    const float* __restrict__ centers, const float* __restrict__ w,
    const float* __restrict__ bias, float* __restrict__ part)
{
  const int bm = blockIdx.x, b = bm / NPATCH, tid = threadIdx.x;
  __shared__ float feat[PPTS * CIN];
  __shared__ float ps[128], pq[128];
  gather_feat(x, knnI, centers, bm, b, tid, feat);
  __syncthreads();
  const int ch = tid & 63, half = tid >> 6;
  float s = 0.f, sq = 0.f;
  for (int p = half * 16; p < half * 16 + 16; ++p) {
    float v = bias[ch];
#pragma unroll
    for (int c = 0; c < CIN; ++c) v += feat[p * CIN + c] * w[ch * CIN + c];
    s += v; sq += v * v;
  }
  ps[tid] = s; pq[tid] = sq;
  __syncthreads();
  if (tid < 64) {
    part[(size_t)bm * 64 + tid]                              = ps[tid] + ps[tid + 64];
    part[(size_t)(BB * NPATCH) * 64 + (size_t)bm * 64 + tid] = pq[tid] + pq[tid + 64];
  }
}

__global__ void k_bnfin(const float* __restrict__ part, int nblk, int C,
                        float* __restrict__ mean, float* __restrict__ inv)
{
  const int ch = threadIdx.x;
  if (ch >= C) return;
  float s = 0.f, sq = 0.f;
  for (int i = 0; i < nblk; ++i) {
    s  += part[(size_t)i * C + ch];
    sq += part[(size_t)nblk * C + (size_t)i * C + ch];
  }
  const float cnt = (float)nblk * (float)PPTS;
  const float m = s / cnt;
  const float v = sq / cnt - m * m;
  mean[ch] = m;
  inv[ch]  = rsqrtf(v + 1e-5f);
}

__device__ inline void conv1_bn_relu(const float* feat, const float* w1c,
                                     const float* b1c, const float* m1,
                                     const float* i1, const float* g1,
                                     const float* bb1, int tid, float* h1)
{
  for (int i = tid; i < PPTS * 64; i += 128) {
    const int p = i >> 6, ch = i & 63;
    float v = b1c[ch];
#pragma unroll
    for (int c = 0; c < CIN; ++c) v += feat[p * CIN + c] * w1c[ch * CIN + c];
    v = (v - m1[ch]) * i1[ch] * g1[ch] + bb1[ch];
    h1[i] = fmaxf(v, 0.f);
  }
}

// conv2 raw stats
__global__ __launch_bounds__(128) void k_pstats2(
    const float* __restrict__ x, const int* __restrict__ knnI,
    const float* __restrict__ centers,
    const float* __restrict__ w1c, const float* __restrict__ b1c,
    const float* __restrict__ m1, const float* __restrict__ i1,
    const float* __restrict__ g1, const float* __restrict__ bb1,
    const float* __restrict__ w2c, const float* __restrict__ b2c,
    float* __restrict__ part)
{
  const int bm = blockIdx.x, b = bm / NPATCH, tid = threadIdx.x;
  __shared__ float feat[PPTS * CIN];
  __shared__ float h1[PPTS * 64];
  gather_feat(x, knnI, centers, bm, b, tid, feat);
  __syncthreads();
  conv1_bn_relu(feat, w1c, b1c, m1, i1, g1, bb1, tid, h1);
  __syncthreads();
  const int ch = tid;          // 128 output channels
  float s = 0.f, sq = 0.f;
  for (int p = 0; p < PPTS; ++p) {
    float v = b2c[ch];
    const float* hp = &h1[p * 64];
#pragma unroll 8
    for (int i = 0; i < 64; ++i) v += hp[i] * w2c[ch * 64 + i];
    s += v; sq += v * v;
  }
  part[(size_t)bm * 128 + ch]                               = s;
  part[(size_t)(BB * NPATCH) * 128 + (size_t)bm * 128 + ch] = sq;
}

// fused conv1->bn->conv2->bn->conv3->maxpool
__global__ __launch_bounds__(128) void k_pfinal(
    const float* __restrict__ x, const int* __restrict__ knnI,
    const float* __restrict__ centers,
    const float* __restrict__ w1c, const float* __restrict__ b1c,
    const float* __restrict__ m1, const float* __restrict__ i1,
    const float* __restrict__ g1, const float* __restrict__ bb1,
    const float* __restrict__ w2c, const float* __restrict__ b2c,
    const float* __restrict__ m2, const float* __restrict__ i2,
    const float* __restrict__ g2, const float* __restrict__ bb2,
    const float* __restrict__ w3c, const float* __restrict__ b3c,
    float* __restrict__ emb)
{
  const int bm = blockIdx.x, b = bm / NPATCH, tid = threadIdx.x;
  __shared__ float feat[PPTS * CIN];
  __shared__ float h1[PPTS * 64];
  __shared__ float h2[PPTS * 128];
  gather_feat(x, knnI, centers, bm, b, tid, feat);
  __syncthreads();
  conv1_bn_relu(feat, w1c, b1c, m1, i1, g1, bb1, tid, h1);
  __syncthreads();
  const int ch = tid;
  for (int p = 0; p < PPTS; ++p) {
    float v = b2c[ch];
    const float* hp = &h1[p * 64];
#pragma unroll 8
    for (int i = 0; i < 64; ++i) v += hp[i] * w2c[ch * 64 + i];
    v = (v - m2[ch]) * i2[ch] * g2[ch] + bb2[ch];
    h2[p * 128 + ch] = fmaxf(v, 0.f);
  }
  __syncthreads();
  for (int r = 0; r < 3; ++r) {
    const int c3 = tid + r * 128;
    float mx = -3.4e38f;
    for (int p = 0; p < PPTS; ++p) {
      float v = b3c[c3];
      const float* hp = &h2[p * 128];
#pragma unroll 8
      for (int i = 0; i < 128; ++i) v += hp[i] * w3c[(size_t)c3 * 128 + i];
      mx = fmaxf(mx, v);
    }
    emb[(size_t)bm * DD + c3] = mx;
  }
}

// position MLP: 3 -> 128 (exact GELU) -> 384, one block per center
__global__ __launch_bounds__(128) void k_posmlp(
    const float* __restrict__ centers,
    const float* __restrict__ w1, const float* __restrict__ b1,
    const float* __restrict__ w2, const float* __restrict__ b2,
    float* __restrict__ out)
{
  const int bm = blockIdx.x, t = threadIdx.x;
  __shared__ float hid[128];
  const float xv = centers[bm * 3 + 0];
  const float yv = centers[bm * 3 + 1];
  const float zv = centers[bm * 3 + 2];
  const float a  = b1[t] + xv * w1[t * 3 + 0] + yv * w1[t * 3 + 1] + zv * w1[t * 3 + 2];
  hid[t] = 0.5f * a * (1.f + erff(a * 0.70710678118654752f));
  __syncthreads();
  for (int r = 0; r < 3; ++r) {
    const int ch = t + r * 128;
    float v = b2[ch];
#pragma unroll 8
    for (int k = 0; k < 128; ++k) v += hid[k] * w2[ch * 128 + k];
    out[(size_t)bm * DD + ch] = v;
  }
}

// LayerNorm over D=384 (96 float4s), one thread per token
__global__ void k_layernorm(const float* __restrict__ x,
                            const float* __restrict__ g,
                            const float* __restrict__ b,
                            float* __restrict__ y, int Mtok)
{
  const int t = blockIdx.x * blockDim.x + threadIdx.x;
  if (t >= Mtok) return;
  const float4* xr = (const float4*)(x + (size_t)t * DD);
  const float4* g4 = (const float4*)g;
  const float4* b4 = (const float4*)b;
  float m = 0.f;
  for (int i = 0; i < DD / 4; ++i) {
    const float4 v = xr[i];
    m += v.x + v.y + v.z + v.w;
  }
  m *= (1.f / DD);
  float var = 0.f;
  for (int i = 0; i < DD / 4; ++i) {
    const float4 v = xr[i];
    const float dx = v.x - m, dy = v.y - m, dz = v.z - m, dw = v.w - m;
    var += dx * dx + dy * dy + dz * dz + dw * dw;
  }
  var *= (1.f / DD);
  const float inv = rsqrtf(var + 1e-5f);
  float4* yr = (float4*)(y + (size_t)t * DD);
  for (int i = 0; i < DD / 4; ++i) {
    const float4 v = xr[i], gg = g4[i], bb = b4[i];
    float4 r;
    r.x = (v.x - m) * inv * gg.x + bb.x;
    r.y = (v.y - m) * inv * gg.y + bb.y;
    r.z = (v.z - m) * inv * gg.z + bb.z;
    r.w = (v.w - m) * inv * gg.w + bb.w;
    yr[i] = r;
  }
}

// attention: one block per (b,h), one thread per query row, scores in
// LDS, all q/k/v traffic as float4 (b128) loads.
__global__ void k_attn(const float* __restrict__ qkv, float* __restrict__ o, int S)
{
  extern __shared__ float sc[];                // S*S floats
  const int b = blockIdx.x / HH, h = blockIdx.x % HH;
  const int q = threadIdx.x;
  const float* base = qkv + (size_t)b * S * (3 * DD);
  const float4* q4  = (const float4*)(base + (size_t)q * (3 * DD) + h * DHEAD);
  float mx = -3.4e38f;
  for (int j = 0; j < S; ++j) {
    const float4* k4 = (const float4*)(base + (size_t)j * (3 * DD) + DD + h * DHEAD);
    float s = 0.f;
#pragma unroll
    for (int d = 0; d < DHEAD / 4; ++d) {
      const float4 qa = q4[d], ka = k4[d];
      s += qa.x * ka.x + qa.y * ka.y + qa.z * ka.z + qa.w * ka.w;
    }
    s *= 0.125f;                               // 1/sqrt(64)
    sc[q * S + j] = s;
    mx = fmaxf(mx, s);
  }
  float sum = 0.f;
  for (int j = 0; j < S; ++j) { const float e = expf(sc[q * S + j] - mx); sc[q * S + j] = e; sum += e; }
  const float inv = 1.f / sum;
  float4 acc[DHEAD / 4];
#pragma unroll
  for (int d = 0; d < DHEAD / 4; ++d) acc[d] = make_float4(0.f, 0.f, 0.f, 0.f);
  for (int j = 0; j < S; ++j) {
    const float p = sc[q * S + j] * inv;
    const float4* v4 = (const float4*)(base + (size_t)j * (3 * DD) + 2 * DD + h * DHEAD);
#pragma unroll
    for (int d = 0; d < DHEAD / 4; ++d) {
      const float4 vv = v4[d];
      acc[d].x += p * vv.x; acc[d].y += p * vv.y;
      acc[d].z += p * vv.z; acc[d].w += p * vv.w;
    }
  }
  float4* orow = (float4*)(o + ((size_t)b * S + q) * DD + h * DHEAD);
#pragma unroll
  for (int d = 0; d < DHEAD / 4; ++d) orow[d] = acc[d];
}

// token assembly kernels
__global__ void k_vis(const float* __restrict__ emb, const float* __restrict__ posF,
                      const int* __restrict__ rand_idx, float* __restrict__ tk)
{
  const int t = blockIdx.x * blockDim.x + threadIdx.x;
  if (t >= BB * NVIS * DD) return;
  const int c = t % DD, s = (t / DD) % NVIS, b = t / (DD * NVIS);
  const int vid = rand_idx[b * NPATCH + NMASK + s];
  const size_t src = ((size_t)b * NPATCH + vid) * DD + c;
  tk[t] = emb[src] + posF[src];
}

__global__ void k_decasm(const float* __restrict__ evis, const float* __restrict__ dposF,
                         const float* __restrict__ mtok, const int* __restrict__ rand_idx,
                         float* __restrict__ tk)
{
  const int t = blockIdx.x * blockDim.x + threadIdx.x;
  if (t >= BB * NPATCH * DD) return;
  const int c = t % DD, r = (t / DD) % NPATCH, b = t / (DD * NPATCH);
  float v;
  if (r < NVIS) {
    const int vid = rand_idx[b * NPATCH + NMASK + r];
    v = evis[((size_t)b * NVIS + r) * DD + c] + dposF[((size_t)b * NPATCH + vid) * DD + c];
  } else {
    const int mi = rand_idx[b * NPATCH + (r - NVIS)];
    v = mtok[c] + dposF[((size_t)b * NPATCH + mi) * DD + c];
  }
  tk[t] = v;
}

__global__ void k_extract(const float* __restrict__ tk, float* __restrict__ dst)
{
  const int t = blockIdx.x * blockDim.x + threadIdx.x;
  if (t >= BB * NMASK * DD) return;
  const int c = t % DD, j = (t / DD) % NMASK, b = t / (DD * NMASK);
  dst[t] = tk[((size_t)b * NPATCH + NVIS + j) * DD + c];
}

__global__ void k_gt(const float* __restrict__ x, const int* __restrict__ knnI,
                     const float* __restrict__ centers, const int* __restrict__ rand_idx,
                     float* __restrict__ out)
{
  const int t = blockIdx.x * blockDim.x + threadIdx.x;
  if (t >= BB * NMASK * PPTS * 3) return;
  const int c = t % 3, p = (t / 3) % PPTS, j = (t / (3 * PPTS)) % NMASK, b = t / (3 * PPTS * NMASK);
  const int mi = rand_idx[b * NPATCH + j];
  const int g  = knnI[((size_t)b * NPATCH + mi) * PPTS + p];
  out[t] = x[((size_t)b * NN + g) * CIN + c] - centers[(b * NPATCH + mi) * 3 + c];
}

// =====================================================================
extern "C" void kernel_launch(void* const* d_in, const int* in_sizes, int n_in,
                              void* d_out, int out_size, void* d_ws, size_t ws_size,
                              hipStream_t stream)
{
  (void)in_sizes; (void)n_in; (void)out_size; (void)ws_size;

  // ---- inputs (setup_inputs order) ----
  const float* x       = (const float*)d_in[0];
  const int*   rand_i  = (const int*)  d_in[1];
  const float* c1w = (const float*)d_in[2],  *c1b = (const float*)d_in[3];
  const float* g1b = (const float*)d_in[4],  *b1b = (const float*)d_in[5];
  const float* c2w = (const float*)d_in[6],  *c2b = (const float*)d_in[7];
  const float* g2b = (const float*)d_in[8],  *b2b = (const float*)d_in[9];
  const float* c3w = (const float*)d_in[10], *c3b = (const float*)d_in[11];
  const float* pw1 = (const float*)d_in[12], *pb1 = (const float*)d_in[13];
  const float* pw2 = (const float*)d_in[14], *pb2 = (const float*)d_in[15];
  const float* dw1 = (const float*)d_in[16], *db1 = (const float*)d_in[17];
  const float* dw2 = (const float*)d_in[18], *db2 = (const float*)d_in[19];
  const float* dew = (const float*)d_in[20], *deb = (const float*)d_in[21];
  const float* mtk = (const float*)d_in[22];
  const float* eWq = (const float*)d_in[23], *ebq = (const float*)d_in[24];
  const float* eWo = (const float*)d_in[25], *ebo = (const float*)d_in[26];
  const float* eg1 = (const float*)d_in[27], *ebg1= (const float*)d_in[28];
  const float* eW1 = (const float*)d_in[29], *eb1 = (const float*)d_in[30];
  const float* eW2 = (const float*)d_in[31], *eb2 = (const float*)d_in[32];
  const float* eg2 = (const float*)d_in[33], *ebg2= (const float*)d_in[34];
  const float* xWq = (const float*)d_in[35], *xbq = (const float*)d_in[36];
  const float* xWo = (const float*)d_in[37], *xbo = (const float*)d_in[38];
  const float* xg1 = (const float*)d_in[39], *xbg1= (const float*)d_in[40];
  const float* xW1 = (const float*)d_in[41], *xb1 = (const float*)d_in[42];
  const float* xW2 = (const float*)d_in[43], *xb2 = (const float*)d_in[44];
  const float* xg2 = (const float*)d_in[45], *xbg2= (const float*)d_in[46];
  const float* hw  = (const float*)d_in[47], *hb  = (const float*)d_in[48];

  // ---- workspace layout (floats, then ints) ----
  float* wf = (float*)d_ws;
  size_t o = 0;
  auto F = [&](size_t n) { float* p = wf + o; o += n; return p; };
  const size_t NPAT = (size_t)BB * NPATCH;
  float* p1part = F(NPAT * 64  * 2);
  float* p2part = F(NPAT * 128 * 2);
  float* bn1m = F(64),  *bn1i = F(64);
  float* bn2m = F(128), *bn2i = F(128);
  float* cent = F(NPAT * 3);
  float* emb  = F(NPAT * DD);
  float* posF = F(NPAT * DD);
  float* dposF= F(NPAT * DD);
  float* tok  = F((size_t)BB * NPATCH * DD);   // up to 8192 tokens
  float* hln  = F((size_t)BB * NPATCH * DD);
  float* obuf = F((size_t)BB * NPATCH * DD);
  float* qkvb = F((size_t)BB * NPATCH * 3 * DD);
  float* ff1  = F((size_t)BB * NPATCH * DFF);
  int* fpsI = (int*)(wf + o); o += NPAT;
  int* knnI = (int*)(wf + o); o += NPAT * PPTS;

  auto gemm = [&](const float* A, const float* W, const float* bias,
                  const float* res, float* Cm, int M, int N, int K, int act) {
    if ((M % 64 == 0) && (N % 64 == 0)) {
      dim3 g(N / 64, M / 64);
      k_gemm_wmma_t<<<g, 256, 0, stream>>>(A, W, bias, res, Cm, M, N, K, act);
    } else {
      dim3 g(N / 16, M / 16);
      k_gemm_wmma<<<g, 32, 0, stream>>>(A, W, bias, res, Cm, M, N, K, act);
    }
  };

  auto layer = [&](float* tk, int Mtok, int S,
                   const float* Wqkv, const float* bqkv,
                   const float* Wo,   const float* bo,
                   const float* g1,   const float* bg1,
                   const float* W1,   const float* bf1,
                   const float* W2,   const float* bf2,
                   const float* g2,   const float* bg2) {
    const int tb = (Mtok + 255) / 256;
    k_layernorm<<<tb, 256, 0, stream>>>(tk, g1, bg1, hln, Mtok);
    gemm(hln, Wqkv, bqkv, nullptr, qkvb, Mtok, 3 * DD, DD, 0);
    k_attn<<<BB * HH, S, (size_t)S * S * sizeof(float), stream>>>(qkvb, obuf, S);
    gemm(obuf, Wo, bo, tk, tk, Mtok, DD, DD, 0);
    k_layernorm<<<tb, 256, 0, stream>>>(tk, g2, bg2, hln, Mtok);
    gemm(hln, W1, bf1, nullptr, ff1, Mtok, DFF, DD, 1);
    gemm(ff1, W2, bf2, tk, tk, Mtok, DD, DFF, 0);
  };

  // ---- 1. FPS + centers ----
  k_fps<<<BB, 256, 0, stream>>>(x, fpsI, cent);
  // ---- 2. kNN for all 128 centers (also serves GT) ----
  k_knn<<<BB * NPATCH, 256, 0, stream>>>(x, cent, knnI);
  // ---- 3. patch embedding (BN stats via deterministic partials) ----
  k_pstats1<<<BB * NPATCH, 128, 0, stream>>>(x, knnI, cent, c1w, c1b, p1part);
  k_bnfin<<<1, 64, 0, stream>>>(p1part, BB * NPATCH, 64, bn1m, bn1i);
  k_pstats2<<<BB * NPATCH, 128, 0, stream>>>(x, knnI, cent, c1w, c1b, bn1m, bn1i,
                                             g1b, b1b, c2w, c2b, p2part);
  k_bnfin<<<1, 128, 0, stream>>>(p2part, BB * NPATCH, 128, bn2m, bn2i);
  k_pfinal<<<BB * NPATCH, 128, 0, stream>>>(x, knnI, cent, c1w, c1b, bn1m, bn1i,
                                            g1b, b1b, c2w, c2b, bn2m, bn2i,
                                            g2b, b2b, c3w, c3b, emb);
  // ---- 4. positional MLPs ----
  k_posmlp<<<BB * NPATCH, 128, 0, stream>>>(cent, pw1, pb1, pw2, pb2, posF);
  k_posmlp<<<BB * NPATCH, 128, 0, stream>>>(cent, dw1, db1, dw2, db2, dposF);
  // ---- 5. visible tokens, encoder (12 layers, 2048 tokens, S=32) ----
  k_vis<<<(BB * NVIS * DD + 255) / 256, 256, 0, stream>>>(emb, posF, rand_i, tok);
  const int MENC = BB * NVIS;
  for (int l = 0; l < ENC_L; ++l) {
    layer(tok, MENC, NVIS,
          eWq + (size_t)l * 3 * DD * DD, ebq + (size_t)l * 3 * DD,
          eWo + (size_t)l * DD * DD,     ebo + (size_t)l * DD,
          eg1 + (size_t)l * DD,          ebg1 + (size_t)l * DD,
          eW1 + (size_t)l * DFF * DD,    eb1 + (size_t)l * DFF,
          eW2 + (size_t)l * DD * DFF,    eb2 + (size_t)l * DD,
          eg2 + (size_t)l * DD,          ebg2 + (size_t)l * DD);
  }
  // ---- 6. demb projection + decoder token assembly ----
  gemm(tok, dew, deb, nullptr, obuf, MENC, DD, DD, 0);
  k_decasm<<<(BB * NPATCH * DD + 255) / 256, 256, 0, stream>>>(obuf, dposF, mtk,
                                                               rand_i, tok);
  // ---- 7. decoder (4 layers, 8192 tokens, S=128) ----
  const int MDEC = BB * NPATCH;
  for (int l = 0; l < DEC_L; ++l) {
    layer(tok, MDEC, NPATCH,
          xWq + (size_t)l * 3 * DD * DD, xbq + (size_t)l * 3 * DD,
          xWo + (size_t)l * DD * DD,     xbo + (size_t)l * DD,
          xg1 + (size_t)l * DD,          xbg1 + (size_t)l * DD,
          xW1 + (size_t)l * DFF * DD,    xb1 + (size_t)l * DFF,
          xW2 + (size_t)l * DD * DFF,    xb2 + (size_t)l * DD,
          xg2 + (size_t)l * DD,          xbg2 + (size_t)l * DD);
  }
  // ---- 8. head -> pred (written straight to d_out) ----
  k_extract<<<(BB * NMASK * DD + 255) / 256, 256, 0, stream>>>(tok, hln);
  gemm(hln, hw, hb, nullptr, (float*)d_out, BB * NMASK, PPTS * 3, DD, 0);
  // ---- 9. GT gather -> d_out second half ----
  const int NG = BB * NMASK * PPTS * 3;
  k_gt<<<(NG + 255) / 256, 256, 0, stream>>>(x, knnI, cent, rand_i,
                                             (float*)d_out + NG);
}
